// SentenceClassifier_64570538328442
// MI455X (gfx1250) — matrix operand order repaired and checked by reading, not compiled, not run on hardware
//
#include <hip/hip_runtime.h>

// Problem constants (from reference)
#define SS 512
#define LL 128
#define EE 256
#define HH 256
#define FAa 128
#define HOPn 4
#define NLn 16
#define G4H 1024   // 4*H
#define KC  512    // E + H (fused GEMM K)
#define D2n 1024   // H*HOP

typedef __attribute__((ext_vector_type(16))) __bf16 v16bf;
typedef __attribute__((ext_vector_type(8)))  __bf16 v8bf;
typedef __attribute__((ext_vector_type(8)))  float  v8f;
typedef __attribute__((ext_vector_type(4)))  float  vf4;

union BFrag { v16bf v; v8bf h[2]; vf4 f[2]; };

__device__ __forceinline__ unsigned short f2bf_bits(float x) {
  union { float f; unsigned u; } a; a.f = x;
  unsigned r = a.u + 0x7FFFu + ((a.u >> 16) & 1u);   // RNE
  return (unsigned short)(r >> 16);
}
__device__ __forceinline__ float sigmoidf(float x) { return 1.0f / (1.0f + __expf(-x)); }

// ---------------- utility kernels ----------------
__global__ void k_zero(float* p, int n) {
  int i = blockIdx.x * blockDim.x + threadIdx.x;
  if (i < n) p[i] = 0.0f;
}

// x = emb[tokens] -> bf16, flat (S*L, E)
__global__ void k_gather(const int* __restrict__ tokens, const float* __restrict__ emb,
                         unsigned short* __restrict__ Xbf, int n) {
  int i = blockIdx.x * blockDim.x + threadIdx.x;
  if (i >= n) return;
  int row = i >> 8;          // /E
  int k   = i & (EE - 1);
  int tok = tokens[row];
  Xbf[i] = f2bf_bits(emb[(size_t)tok * EE + k]);
}

// Wcat[dir][n][k] : k<E -> Wih[n][k], else Whh[n][k-E]  (bf16, row-major KC)
__global__ void k_prep_w(const float* __restrict__ WihF, const float* __restrict__ WhhF,
                         const float* __restrict__ WihB, const float* __restrict__ WhhB,
                         unsigned short* __restrict__ Wcat, int n) {
  int i = blockIdx.x * blockDim.x + threadIdx.x;
  if (i >= n) return;
  int dir = i / (G4H * KC);
  int r   = i % (G4H * KC);
  int row = r / KC;
  int k   = r % KC;
  const float* Wih = dir ? WihB : WihF;
  const float* Whh = dir ? WhhB : WhhF;
  float v = (k < EE) ? Wih[(size_t)row * EE + k] : Whh[(size_t)row * HH + (k - EE)];
  Wcat[i] = f2bf_bits(v);
}

__global__ void k_prep_b(const float* bihF, const float* bhhF,
                         const float* bihB, const float* bhhB, float* bias) {
  int i = blockIdx.x * blockDim.x + threadIdx.x;
  if (i >= 2 * G4H) return;
  int dir = i / G4H, n = i % G4H;
  bias[i] = dir ? (bihB[n] + bhhB[n]) : (bihF[n] + bhhF[n]);
}

// ---------------- fused LSTM step: [x_t | h] @ Wcat.T + bias -> gates -> c,h ----------------
#define BM 32
#define BN 32
#define LDA 48   // LDS row stride (bf16 elems), multiple of 8 for 16B alignment

__global__ __launch_bounds__(128) void k_lstm_step(
    const unsigned short* __restrict__ Xbf,   // (S*L, E) bf16
    const unsigned short* __restrict__ Wcat,  // [2][G4H][KC] bf16
    const float* __restrict__ bias,           // [2][G4H]
    const float* __restrict__ hIn,            // [2][S][H]
    float* __restrict__ hOut,                 // [2][S][H]
    float* __restrict__ cSt,                  // [2][S][H]
    float* __restrict__ encF,                 // (S, L, H)
    float* __restrict__ encB,
    int t) {
  __shared__ unsigned short Abf[BM * LDA];
  __shared__ float Gsh[4][BM][BN + 1];

  const int dir   = blockIdx.z;
  const int sBase = blockIdx.x * BM;
  const int nBase = blockIdx.y * BN;     // column within each gate's 256
  const int tid   = threadIdx.x;
  const int wave  = tid >> 5;            // gate index 0..3 (i,f,g,o)
  const int lane  = tid & 31;

  const unsigned short* W = Wcat + (size_t)dir * G4H * KC;
  const float* bi = bias + dir * G4H;
  const float* hI = hIn  + (size_t)dir * SS * HH;
  float* hO  = hOut + (size_t)dir * SS * HH;
  float* cS  = cSt  + (size_t)dir * SS * HH;
  float* enc = dir ? encB : encF;
  const int l = dir ? (LL - 1 - t) : t;

  v8f acc[2][2] = {};

  const int mS   = tid >> 2;             // 0..31  (staging row)
  const int koff = (tid & 3) * 8;        // 0,8,16,24
  const int sRow = sBase + mS;

  for (int kb = 0; kb < KC; kb += 32) {
    __syncthreads();
    // ---- stage A tile (32 rows x 32 k) as bf16 into LDS ----
    if (kb < EE) {
      const unsigned short* src = Xbf + (((size_t)sRow * LL + t) * EE + kb + koff);
      *(vf4*)&Abf[mS * LDA + koff] = *(const vf4*)src;
    } else {
      const float* src = hI + (size_t)sRow * HH + (kb - EE) + koff;
      alignas(16) unsigned short tmp[8];
#pragma unroll
      for (int e = 0; e < 8; ++e) tmp[e] = f2bf_bits(src[e]);
      *(vf4*)&Abf[mS * LDA + koff] = *(const vf4*)tmp;
    }
    __syncthreads();

    // ---- A fragments from LDS (ISA 16-bit A 16x32 layout) ----
    BFrag a0, a1;
    {
      const int mm    = lane & 15;
      const int base0 = (lane < 16) ? 0 : 8;
      a0.f[0] = *(const vf4*)&Abf[(mm)      * LDA + base0];
      a0.f[1] = *(const vf4*)&Abf[(mm)      * LDA + base0 + 16];
      a1.f[0] = *(const vf4*)&Abf[(mm + 16) * LDA + base0];
      a1.f[1] = *(const vf4*)&Abf[(mm + 16) * LDA + base0 + 16];
    }
    // ---- B fragments straight from global (L2-hot weights) ----
    BFrag b0, b1;
    {
      const int nn    = lane & 15;
      const int khalf = (lane >> 4) * 8;
      const unsigned short* w0 = W + ((size_t)(wave * HH + nBase +      nn) * KC + kb + khalf);
      const unsigned short* w1 = W + ((size_t)(wave * HH + nBase + 16 + nn) * KC + kb + khalf);
      b0.f[0] = *(const vf4*)(w0);
      b0.f[1] = *(const vf4*)(w0 + 16);
      b1.f[0] = *(const vf4*)(w1);
      b1.f[1] = *(const vf4*)(w1 + 16);
    }
    acc[0][0] = __builtin_amdgcn_wmma_f32_16x16x32_bf16(false, a0.v, false, b0.v, (short)0, acc[0][0], false, false);
    acc[0][1] = __builtin_amdgcn_wmma_f32_16x16x32_bf16(false, a0.v, false, b1.v, (short)0, acc[0][1], false, false);
    acc[1][0] = __builtin_amdgcn_wmma_f32_16x16x32_bf16(false, a1.v, false, b0.v, (short)0, acc[1][0], false, false);
    acc[1][1] = __builtin_amdgcn_wmma_f32_16x16x32_bf16(false, a1.v, false, b1.v, (short)0, acc[1][1], false, false);
  }

  // ---- exchange accumulators (each wave owns one gate) ----
  __syncthreads();
#pragma unroll
  for (int mi = 0; mi < 2; ++mi)
#pragma unroll
    for (int ni = 0; ni < 2; ++ni) {
      const int mT = mi * 16 + ((lane >> 4) << 3);
      const int nT = ni * 16 + (lane & 15);
#pragma unroll
      for (int r = 0; r < 8; ++r)
        Gsh[wave][mT + r][nT] = acc[mi][ni][r];
    }
  __syncthreads();

  // ---- fused LSTM cell update for this (s,j) tile ----
#pragma unroll
  for (int e = 0; e < 8; ++e) {
    int flat = tid + e * 128;
    int m = flat >> 5, j = flat & 31;
    int ss = sBase + m, jg = nBase + j;
    float gi = Gsh[0][m][j] + bi[jg];
    float gf = Gsh[1][m][j] + bi[HH + jg];
    float gg = Gsh[2][m][j] + bi[2 * HH + jg];
    float go = Gsh[3][m][j] + bi[3 * HH + jg];
    size_t sidx = (size_t)ss * HH + jg;
    float c0 = cS[sidx];
    float cn = sigmoidf(gf) * c0 + sigmoidf(gi) * tanhf(gg);
    float hn = sigmoidf(go) * tanhf(cn);
    cS[sidx] = cn;
    hO[sidx] = hn;
    enc[((size_t)ss * LL + l) * HH + jg] = hn;
  }
}

// ---------------- attention (algebraically reduced) ----------------
__global__ void k_add(float* __restrict__ a, const float* __restrict__ b, int n) {
  int i = blockIdx.x * blockDim.x + threadIdx.x;
  if (i < n) a[i] += b[i];
}

// Mred[d][h] = sum_f attW[f][d]*ctx[h][f];  b4[h] = sum_f attb[f]*ctx[h][f]
__global__ void k_mred(const float* __restrict__ attW, const float* __restrict__ attb,
                       const float* __restrict__ ctx, float* Mred, float* b4) {
  int d = threadIdx.x;  // 256
  for (int h = 0; h < HOPn; ++h) {
    float a = 0.f;
    for (int f = 0; f < FAa; ++f) a += attW[(size_t)f * HH + d] * ctx[h * FAa + f];
    Mred[d * HOPn + h] = a;
  }
  if (d < HOPn) {
    float a = 0.f;
    for (int f = 0; f < FAa; ++f) a += attb[f] * ctx[d * FAa + f];
    b4[d] = a;
  }
}

__global__ void k_scores(const float* __restrict__ enc, const float* __restrict__ Mred,
                         const float* __restrict__ b4, float* __restrict__ scores) {
  int row = blockIdx.x * blockDim.x + threadIdx.x;  // s*L + l
  if (row >= SS * LL) return;
  const float* er = enc + (size_t)row * HH;
  float s0 = b4[0], s1 = b4[1], s2 = b4[2], s3 = b4[3];
  for (int d = 0; d < HH; ++d) {
    float ev = er[d];
    s0 += ev * Mred[d * 4 + 0];
    s1 += ev * Mred[d * 4 + 1];
    s2 += ev * Mred[d * 4 + 2];
    s3 += ev * Mred[d * 4 + 3];
  }
  float* o = scores + (size_t)row * HOPn;
  o[0] = s0; o[1] = s1; o[2] = s2; o[3] = s3;
}

// softmax over l (per s, per hop) — wave32 shuffle reductions
__global__ void k_softmax(const float* __restrict__ scores, float* __restrict__ attn) {
  int s = blockIdx.x, lt = threadIdx.x;  // 32 threads, 4 l's each
  float v[4][HOPn];
#pragma unroll
  for (int q = 0; q < 4; ++q) {
    const float* p = scores + ((size_t)s * LL + lt + 32 * q) * HOPn;
#pragma unroll
    for (int h = 0; h < HOPn; ++h) v[q][h] = p[h];
  }
#pragma unroll
  for (int h = 0; h < HOPn; ++h) {
    float mx = v[0][h];
#pragma unroll
    for (int q = 1; q < 4; ++q) mx = fmaxf(mx, v[q][h]);
    for (int off = 16; off > 0; off >>= 1) mx = fmaxf(mx, __shfl_xor(mx, off));
    float sum = 0.f;
#pragma unroll
    for (int q = 0; q < 4; ++q) { v[q][h] = __expf(v[q][h] - mx); sum += v[q][h]; }
    for (int off = 16; off > 0; off >>= 1) sum += __shfl_xor(sum, off);
    float inv = 1.0f / sum;
#pragma unroll
    for (int q = 0; q < 4; ++q) v[q][h] *= inv;
  }
#pragma unroll
  for (int q = 0; q < 4; ++q) {
    float* p = attn + ((size_t)s * LL + lt + 32 * q) * HOPn;
#pragma unroll
    for (int h = 0; h < HOPn; ++h) p[h] = v[q][h];
  }
}

// sent[s][d][h] = sum_l enc[s][l][d] * attn[s][l][h]
__global__ void k_sent(const float* __restrict__ enc, const float* __restrict__ attn,
                       float* __restrict__ sent) {
  __shared__ float at[LL][HOPn];
  int s = blockIdx.x, d = threadIdx.x;  // 256 threads
  for (int i = threadIdx.x; i < LL * HOPn; i += blockDim.x)
    at[i / HOPn][i % HOPn] = attn[(size_t)s * LL * HOPn + i];
  __syncthreads();
  float a0 = 0, a1 = 0, a2 = 0, a3 = 0;
  for (int l = 0; l < LL; ++l) {
    float ev = enc[((size_t)s * LL + l) * HH + d];
    a0 += ev * at[l][0]; a1 += ev * at[l][1]; a2 += ev * at[l][2]; a3 += ev * at[l][3];
  }
  float* o = sent + (size_t)s * D2n + d * HOPn;
  o[0] = a0; o[1] = a1; o[2] = a2; o[3] = a3;
}

// ---------------- sentence-level LSTM ----------------
// P[dir][s][g] = sum_k sent[s][k] * sWih_dir[g][k]
__global__ void k_sentproj(const float* __restrict__ sent, const float* __restrict__ sWihF,
                           const float* __restrict__ sWihB, float* __restrict__ P) {
  int i = blockIdx.x * blockDim.x + threadIdx.x;
  if (i >= 2 * SS * 16) return;
  int dir = i / (SS * 16);
  int r = i % (SS * 16);
  int s = r / 16, g = r % 16;
  const float* W = dir ? sWihB : sWihF;
  const float* x = sent + (size_t)s * D2n;
  const float* w = W + (size_t)g * D2n;
  float a = 0.f;
  for (int k = 0; k < D2n; ++k) a += x[k] * w[k];
  P[i] = a;
}

__global__ void k_sent_scan(const float* __restrict__ P,
                            const float* __restrict__ sWhhF, const float* __restrict__ sWhhB,
                            const float* sbihF, const float* sbhhF,
                            const float* sbihB, const float* sbhhB,
                            float* __restrict__ sencFB) {  // [2][S][HOP]
  __shared__ float hsh[2][HOPn], csh[2][HOPn], gsh[2][16];
  int tid = threadIdx.x;        // 32
  int dir = tid >> 4, g = tid & 15;
  if (g < HOPn) { hsh[dir][g] = 0.f; csh[dir][g] = 0.f; }
  const float* Whh = dir ? sWhhB : sWhhF;
  float bg = dir ? (sbihB[g] + sbhhB[g]) : (sbihF[g] + sbhhF[g]);
  const float* Pd = P + (size_t)dir * SS * 16;
  float* od = sencFB + (size_t)dir * SS * HOPn;
  __syncthreads();
  for (int t = 0; t < SS; ++t) {
    int si = dir ? (SS - 1 - t) : t;
    float pre = Pd[si * 16 + g] + bg;
#pragma unroll
    for (int j = 0; j < HOPn; ++j) pre += hsh[dir][j] * Whh[g * HOPn + j];
    gsh[dir][g] = pre;
    __syncthreads();
    if (g < HOPn) {
      int j = g;
      float cn = sigmoidf(gsh[dir][4 + j]) * csh[dir][j] + sigmoidf(gsh[dir][j]) * tanhf(gsh[dir][8 + j]);
      float hn = sigmoidf(gsh[dir][12 + j]) * tanhf(cn);
      csh[dir][j] = cn; hsh[dir][j] = hn;
      od[si * HOPn + j] = hn;
    }
    __syncthreads();
  }
}

// pf[t][n] = sum_j (sF+sB)[t][j]*linW[n][j] + linb[n]; then Viterbi max-plus scan
__global__ void k_viterbi(const float* __restrict__ sencFB, const float* __restrict__ linW,
                          const float* __restrict__ linb, const float* __restrict__ trans,
                          float* __restrict__ out) {
  __shared__ float v[NLn];
  int n = threadIdx.x;  // 32, only <16 active
  if (n < NLn) {
    float p = linb[n];
#pragma unroll
    for (int j = 0; j < HOPn; ++j)
      p += (sencFB[j] + sencFB[(size_t)SS * HOPn + j]) * linW[n * HOPn + j];
    v[n] = p; out[n] = p;
  }
  __syncthreads();
  for (int t = 1; t < SS; ++t) {
    float vn = 0.f;
    if (n < NLn) {
      float p = linb[n];
      const float* sf = sencFB + (size_t)t * HOPn;
      const float* sb = sencFB + (size_t)(SS + t) * HOPn;
#pragma unroll
      for (int j = 0; j < HOPn; ++j) p += (sf[j] + sb[j]) * linW[n * HOPn + j];
      float best = -3.4e38f;
      for (int i = 0; i < NLn; ++i) best = fmaxf(best, v[i] + trans[i * NLn + n]);
      vn = p + best;
    }
    __syncthreads();
    if (n < NLn) { v[n] = vn; out[(size_t)t * NLn + n] = vn; }
    __syncthreads();
  }
}

// ---------------- host side ----------------
extern "C" void kernel_launch(void* const* d_in, const int* in_sizes, int n_in,
                              void* d_out, int out_size, void* d_ws, size_t ws_size,
                              hipStream_t stream) {
  const int*   tokens = (const int*)  d_in[0];
  const float* emb    = (const float*)d_in[1];
  const float* tWih_f = (const float*)d_in[2];
  const float* tWhh_f = (const float*)d_in[3];
  const float* tbih_f = (const float*)d_in[4];
  const float* tbhh_f = (const float*)d_in[5];
  const float* tWih_b = (const float*)d_in[6];
  const float* tWhh_b = (const float*)d_in[7];
  const float* tbih_b = (const float*)d_in[8];
  const float* tbhh_b = (const float*)d_in[9];
  const float* attW   = (const float*)d_in[10];
  const float* attb   = (const float*)d_in[11];
  const float* ctx    = (const float*)d_in[12];
  const float* sWih_f = (const float*)d_in[13];
  const float* sWhh_f = (const float*)d_in[14];
  const float* sbih_f = (const float*)d_in[15];
  const float* sbhh_f = (const float*)d_in[16];
  const float* sWih_b = (const float*)d_in[17];
  const float* sWhh_b = (const float*)d_in[18];
  const float* sbih_b = (const float*)d_in[19];
  const float* sbhh_b = (const float*)d_in[20];
  const float* linW   = (const float*)d_in[21];
  const float* linb   = (const float*)d_in[22];
  const float* trans  = (const float*)d_in[23];
  float* out = (float*)d_out;

  // workspace carve-up (~170 MB)
  char* w = (char*)d_ws;
  auto alloc = [&](size_t bytes) -> char* {
    char* p = w;
    w += (bytes + 255) & ~(size_t)255;
    return p;
  };
  unsigned short* Xbf  = (unsigned short*)alloc((size_t)SS * LL * EE * 2);
  unsigned short* Wcat = (unsigned short*)alloc((size_t)2 * G4H * KC * 2);
  float* bias  = (float*)alloc((size_t)2 * G4H * 4);
  float* hbuf  = (float*)alloc((size_t)2 * 2 * SS * HH * 4);  // [buf][dir][S][H]
  float* cbuf  = (float*)alloc((size_t)2 * SS * HH * 4);      // [dir][S][H]
  float* encF  = (float*)alloc((size_t)SS * LL * HH * 4);
  float* encB  = (float*)alloc((size_t)SS * LL * HH * 4);
  float* Mred  = (float*)alloc((size_t)HH * HOPn * 4);
  float* b4    = (float*)alloc((size_t)HOPn * 4);
  float* scores= (float*)alloc((size_t)SS * LL * HOPn * 4);
  float* attn  = (float*)alloc((size_t)SS * LL * HOPn * 4);
  float* sent  = (float*)alloc((size_t)SS * D2n * 4);
  float* P     = (float*)alloc((size_t)2 * SS * 16 * 4);
  float* sencFB= (float*)alloc((size_t)2 * SS * HOPn * 4);

  // init recurrent state
  k_zero<<<(2 * 2 * SS * HH + 255) / 256, 256, 0, stream>>>(hbuf, 2 * 2 * SS * HH);
  k_zero<<<(2 * SS * HH + 255) / 256, 256, 0, stream>>>(cbuf, 2 * SS * HH);

  // embed + weight packing
  int nx = SS * LL * EE;
  k_gather<<<(nx + 255) / 256, 256, 0, stream>>>(tokens, emb, Xbf, nx);
  int nw = 2 * G4H * KC;
  k_prep_w<<<(nw + 255) / 256, 256, 0, stream>>>(tWih_f, tWhh_f, tWih_b, tWhh_b, Wcat, nw);
  k_prep_b<<<(2 * G4H + 255) / 256, 256, 0, stream>>>(tbih_f, tbhh_f, tbih_b, tbhh_b, bias);

  // token-level BiLSTM: 128 sequential fused GEMM+cell steps, both directions per launch
  dim3 gstep(SS / BM, HH / BN, 2);
  const size_t hstride = (size_t)2 * SS * HH;
  for (int t = 0; t < LL; ++t) {
    float* hIn  = hbuf + (size_t)(t & 1) * hstride;
    float* hOut = hbuf + (size_t)((t + 1) & 1) * hstride;
    k_lstm_step<<<gstep, 128, 0, stream>>>(Xbf, Wcat, bias, hIn, hOut, cbuf, encF, encB, t);
  }

  // enc = encF + encB (in place into encF)
  int ne = SS * LL * HH;
  k_add<<<(ne + 255) / 256, 256, 0, stream>>>(encF, encB, ne);

  // attention (reduced): scores = enc @ Mred + b4; softmax over L; sent einsum
  k_mred<<<1, 256, 0, stream>>>(attW, attb, ctx, Mred, b4);
  k_scores<<<(SS * LL + 255) / 256, 256, 0, stream>>>(encF, Mred, b4, scores);
  k_softmax<<<SS, 32, 0, stream>>>(scores, attn);
  k_sent<<<SS, 256, 0, stream>>>(encF, attn, sent);

  // sentence-level BiLSTM
  k_sentproj<<<(2 * SS * 16 + 255) / 256, 256, 0, stream>>>(sent, sWih_f, sWih_b, P);
  k_sent_scan<<<1, 32, 0, stream>>>(P, sWhh_f, sWhh_b, sbih_f, sbhh_f, sbih_b, sbhh_b, sencFB);

  // final linear + Viterbi
  k_viterbi<<<1, 32, 0, stream>>>(sencFB, linW, linb, trans, out);
}
